// TemporalAttnLayer_83270825934857
// MI455X (gfx1250) — matrix operand order repaired and checked by reading, not compiled, not run on hardware
//
#include <hip/hip_runtime.h>

#define B_DIM 256
#define T_DIM 64
#define M_DIM 256
#define H_DIM 256

typedef __attribute__((ext_vector_type(16))) __bf16 v16bf;
typedef __attribute__((ext_vector_type(8)))  __bf16 v8bf;
typedef __attribute__((ext_vector_type(8)))  float  v8f;

__device__ __forceinline__ v8f wmma_bf16f32(v16bf a, v16bf b, v8f c) {
  // emits v_wmma_f32_16x16x32_bf16
  return __builtin_amdgcn_wmma_f32_16x16x32_bf16(false, a, false, b, (short)0, c,
                                                 false, false);
}

// A-matrix (16x32 bf16) fragment: lanes 0-15 -> row=lane, K chunks {0..7,16..23};
// lanes 16-31 -> row=lane-16, K chunks {8..15,24..31}  (ISA 7.12.2 layout)
__device__ __forceinline__ v16bf load_a_frag(const __bf16* base, int ldk, int kbase) {
  const int lane = threadIdx.x & 31;
  const __bf16* p = base + (size_t)(lane & 15) * ldk + kbase + ((lane >> 4) << 3);
  v8bf lo = *(const v8bf*)p;
  v8bf hi = *(const v8bf*)(p + 16);
  return __builtin_shufflevector(lo, hi, 0, 1, 2, 3, 4, 5, 6, 7,
                                 8, 9, 10, 11, 12, 13, 14, 15);
}

// B-matrix (32x16 bf16) fragment from packed tile: lane holds 16 contiguous bf16
__device__ __forceinline__ v16bf load_b_frag(const __bf16* tile) {
  return *(const v16bf*)(tile + (threadIdx.x & 31) * 16);
}

__device__ __forceinline__ float sigm(float x) { return 1.0f / (1.0f + __expf(-x)); }

// ---------------------------------------------------------------------------
// Pack an (N x K) row-major fp32 weight W (used as x @ W^T) into bf16 WMMA
// B-tiles: tile (nt,kt) of 512 bf16, lane l holds B[k=kt*32+(l>=16?8:0)+(j&7)+(j>=8?16:0)][n=nt*16+l%16]
// ---------------------------------------------------------------------------
__global__ __launch_bounds__(256) void pack_b_kernel(const float* __restrict__ W,
                                                     __bf16* __restrict__ out,
                                                     int N, int K) {
  int idx = blockIdx.x * 256 + threadIdx.x;
  if (idx >= N * K) return;
  int j    = idx & 15;
  int lane = (idx >> 4) & 31;
  int tile = idx >> 9;
  int ktiles = K >> 5;
  int kt = tile % ktiles;
  int nt = tile / ktiles;
  int n = nt * 16 + (lane & 15);
  int k = kt * 32 + ((lane >> 4) << 3) + (j & 7) + ((j >> 3) << 4);
  out[idx] = (__bf16)W[(size_t)n * K + k];
}

// enc_inputs (T,B,M) fp32 -> enc_bf (B,T,M) bf16 (A-matrix operand of y1 GEMM)
__global__ __launch_bounds__(256) void enc_convert_kernel(const float* __restrict__ enc,
                                                          __bf16* __restrict__ enc_bf) {
  int idx = blockIdx.x * 256 + threadIdx.x;           // over B*T*M = 4M
  int m = idx & (M_DIM - 1);
  int t = (idx >> 8) & (T_DIM - 1);
  int b = idx >> 14;
  enc_bf[idx] = (__bf16)enc[((size_t)t * B_DIM + b) * M_DIM + m];
}

// y1[bt,n] = sum_m enc[bt,m] * U_d[n,m]   (16384 x 256, K=256)
// one 16x16 output tile per wave, 8 waves per block
__global__ __launch_bounds__(256) void y1_gemm_kernel(const __bf16* __restrict__ enc_bf,
                                                      const __bf16* __restrict__ Ud_p,
                                                      float* __restrict__ y1) {
  int wave = blockIdx.x * 8 + (threadIdx.x >> 5);
  int lane = threadIdx.x & 31;
  int nt  = wave & 15;        // N tile (M_DIM/16)
  int btt = wave >> 4;        // B*T tile
  const __bf16* arow = enc_bf + (size_t)btt * 16 * M_DIM;
  v8f acc = {};
  for (int kt = 0; kt < 8; ++kt) {
    v16bf a = load_a_frag(arow, M_DIM, kt * 32);
    v16bf b = load_b_frag(Ud_p + ((nt * 8 + kt) << 9));
    acc = wmma_bf16f32(a, b, acc);
  }
  int col = nt * 16 + (lane & 15);
  int r0  = btt * 16 + ((lane >> 4) << 3);
  for (int r = 0; r < 8; ++r)
    y1[(size_t)(r0 + r) * M_DIM + col] = acc[r];
}

// ---------------------------------------------------------------------------
// Persistent recurrence: 16 workgroups x 512 threads (16 waves = 4/SIMD32).
// Each WG owns 16 batch rows for all 64 steps; carry lives in LDS; weights
// stream from L2. Every wave owns one 16-col N-tile per GEMM stage.
// ---------------------------------------------------------------------------
__global__ __launch_bounds__(512) void recurrence_kernel(
    const float* __restrict__ enc,      // (T,B,M)
    const float* __restrict__ yin,      // (B,T,1)
    const float* __restrict__ y1,       // (B*T, M)
    const __bf16* __restrict__ Wd_p,    // packed N=256,K=512
    const __bf16* __restrict__ Whh_p,   // packed N=1024,K=256
    const __bf16* __restrict__ Wy_p,    // packed N=256,K=512
    const float* __restrict__ b_Wd,
    const float* __restrict__ v_d,
    const float* __restrict__ w_tilda,  // M+1
    const float* __restrict__ b_wt,
    const float* __restrict__ W_ih,     // (4H)
    const float* __restrict__ b_ih,
    const float* __restrict__ b_hh,
    const float* __restrict__ b_Wy,
    const float* __restrict__ v_y,
    const float* __restrict__ b_vy,
    float* __restrict__ out) {

  __shared__ float  sp_s[16][M_DIM];        // cell state (fp32)
  __shared__ float  c_s [16][M_DIM];        // context
  __shared__ float  x1_s[16][M_DIM];        // attention pre-activation / final h
  __shared__ __bf16 a_bf[16][2 * M_DIM];    // [d | sp] bf16 A-matrix operand
  __shared__ float  l_s [16][T_DIM];        // scores -> beta
  __shared__ float  red [16][16];           // dot-product reduction scratch
  __shared__ float  ytil_s[16];

  const int tid    = threadIdx.x;
  const int wave   = tid >> 5;              // 0..15
  const int lane   = tid & 31;
  const int lcol   = lane & 15;
  const int rowoff = (lane >> 4) << 3;
  const int b0     = blockIdx.x * 16;

  for (int i = tid; i < 16 * M_DIM; i += 512) sp_s[i >> 8][i & 255] = 0.0f;
  for (int i = tid; i < 16 * 2 * M_DIM; i += 512) a_bf[i >> 9][i & 511] = (__bf16)0.0f;

  for (int t = 0; t < T_DIM; ++t) {
    __syncthreads();
    // (a) x1 = [d|sp] @ W_d^T + b_Wd : one 16-col tile per wave, K=512
    {
      int nt = wave;
      v8f acc = {};
      for (int kt = 0; kt < 16; ++kt) {
        v16bf a  = load_a_frag(&a_bf[0][0], 2 * M_DIM, kt * 32);
        v16bf bm = load_b_frag(Wd_p + ((nt * 16 + kt) << 9));
        acc = wmma_bf16f32(a, bm, acc);
      }
      int col = nt * 16 + lcol;
      float bias = b_Wd[col];
      for (int r = 0; r < 8; ++r) x1_s[r + rowoff][col] = acc[r] + bias;
    }
    __syncthreads();
    // (b) l[b,tt] = sum_m tanh(x1[b,m] + y1[b,tt,m]) * v_d[m] : 2 pairs/thread
    for (int p = 0; p < 2; ++p) {
      int pair = p * 512 + tid;          // 16*64 = 1024 pairs
      int b  = pair >> 6;
      int tt = pair & 63;
      const float4* yv = (const float4*)(y1 + ((size_t)(b0 + b) * T_DIM + tt) * M_DIM);
      const float4* xv = (const float4*)(&x1_s[b][0]);
      const float4* vv = (const float4*)v_d;
      float s = 0.0f;
      for (int m = 0; m < M_DIM / 4; ++m) {
        float4 a = yv[m], x = xv[m], v = vv[m];
        s += tanhf(x.x + a.x) * v.x + tanhf(x.y + a.y) * v.y +
             tanhf(x.z + a.z) * v.z + tanhf(x.w + a.w) * v.w;
      }
      l_s[b][tt] = s;
    }
    __syncthreads();
    // softmax over T per batch row
    if (tid < 16) {
      float mx = -1e30f;
      for (int i = 0; i < T_DIM; ++i) mx = fmaxf(mx, l_s[tid][i]);
      float sum = 0.0f;
      for (int i = 0; i < T_DIM; ++i) {
        float e = __expf(l_s[tid][i] - mx);
        l_s[tid][i] = e;
        sum += e;
      }
      float inv = 1.0f / sum;
      for (int i = 0; i < T_DIM; ++i) l_s[tid][i] *= inv;
    }
    __syncthreads();
    // (c) c[b,m] = sum_t beta[b,t] * enc[t,b,m] : 8 outputs/thread
    {
      int b  = tid >> 5;
      int m0 = (tid & 31) << 3;
      float acc[8];
      for (int i = 0; i < 8; ++i) acc[i] = 0.0f;
      for (int t2 = 0; t2 < T_DIM; ++t2) {
        float w = l_s[b][t2];
        const float4* ev =
            (const float4*)(enc + ((size_t)t2 * B_DIM + b0 + b) * M_DIM + m0);
        for (int i = 0; i < 2; ++i) {
          float4 e = ev[i];
          acc[i * 4 + 0] += w * e.x;
          acc[i * 4 + 1] += w * e.y;
          acc[i * 4 + 2] += w * e.z;
          acc[i * 4 + 3] += w * e.w;
        }
      }
      for (int i = 0; i < 8; ++i) c_s[b][m0 + i] = acc[i];
    }
    __syncthreads();
    // (d) y_til[b] = [c|y_t] . w_tilda + b_wt : 16-way parallel dot + reduce
    if (tid < 256) {
      int b   = tid >> 4;
      int seg = tid & 15;
      float s = 0.0f;
      for (int i = 0; i < 16; ++i) {
        int m = seg * 16 + i;
        s += c_s[b][m] * w_tilda[m];
      }
      red[b][seg] = s;
    }
    __syncthreads();
    if (tid < 16) {
      float s = b_wt[0] + yin[(size_t)(b0 + tid) * T_DIM + t] * w_tilda[M_DIM];
      for (int i = 0; i < 16; ++i) s += red[tid][i];
      ytil_s[tid] = s;
    }
    __syncthreads();
    // (e) gates = y_til*W_ih^T + d@W_hh^T + biases ; LSTM update.
    // Wave w owns h-cols [w*16, w*16+16): 4 gate tiles share one A fragment.
    float spn[8], dn[8];
    {
      int nt = wave;                 // h tile 0..15
      v8f ai = {}, af = {}, ag = {}, ao = {};
      for (int kt = 0; kt < 8; ++kt) {
        v16bf a = load_a_frag(&a_bf[0][0], 2 * M_DIM, kt * 32);  // d half only
        ai = wmma_bf16f32(a, load_b_frag(Whh_p + ((( 0 + nt) * 8 + kt) << 9)), ai);
        af = wmma_bf16f32(a, load_b_frag(Whh_p + (((16 + nt) * 8 + kt) << 9)), af);
        ag = wmma_bf16f32(a, load_b_frag(Whh_p + (((32 + nt) * 8 + kt) << 9)), ag);
        ao = wmma_bf16f32(a, load_b_frag(Whh_p + (((48 + nt) * 8 + kt) << 9)), ao);
      }
      int col = nt * 16 + lcol;
      float wii = W_ih[col],             bii = b_ih[col]             + b_hh[col];
      float wif = W_ih[H_DIM + col],     bif = b_ih[H_DIM + col]     + b_hh[H_DIM + col];
      float wig = W_ih[2 * H_DIM + col], big = b_ih[2 * H_DIM + col] + b_hh[2 * H_DIM + col];
      float wio = W_ih[3 * H_DIM + col], bio = b_ih[3 * H_DIM + col] + b_hh[3 * H_DIM + col];
      for (int r = 0; r < 8; ++r) {
        int b = r + rowoff;
        float yt = ytil_s[b];
        float gi = ai[r] + yt * wii + bii;
        float gf = af[r] + yt * wif + bif;
        float gg = ag[r] + yt * wig + big;
        float go = ao[r] + yt * wio + bio;
        float sn = sigm(gf) * sp_s[b][col] + sigm(gi) * tanhf(gg);
        spn[r] = sn;
        dn[r]  = sigm(go) * tanhf(sn);
      }
    }
    __syncthreads();   // everyone done reading old carry before overwrite
    {
      int col = wave * 16 + lcol;
      for (int r = 0; r < 8; ++r) {
        int b = r + rowoff;
        sp_s[b][col]         = spn[r];
        a_bf[b][col]         = (__bf16)dn[r];
        a_bf[b][M_DIM + col] = (__bf16)spn[r];
      }
    }
  }
  __syncthreads();
  // A := [d | c] for output head
  for (int i = tid; i < 16 * M_DIM; i += 512) {
    int b = i >> 8, m = i & 255;
    a_bf[b][M_DIM + m] = (__bf16)c_s[b][m];
  }
  __syncthreads();
  // h = [d|c] @ W_y^T + b_Wy  (K=512) -> x1_s ; one tile per wave
  {
    int nt = wave;
    v8f acc = {};
    for (int kt = 0; kt < 16; ++kt) {
      v16bf a = load_a_frag(&a_bf[0][0], 2 * M_DIM, kt * 32);
      acc = wmma_bf16f32(a, load_b_frag(Wy_p + ((nt * 16 + kt) << 9)), acc);
    }
    int col = nt * 16 + lcol;
    float bias = b_Wy[col];
    for (int r = 0; r < 8; ++r) x1_s[r + rowoff][col] = acc[r] + bias;
  }
  __syncthreads();
  // out[b] = h[b,:] . v_y + b_vy : parallel dot + reduce
  if (tid < 256) {
    int b   = tid >> 4;
    int seg = tid & 15;
    float s = 0.0f;
    for (int i = 0; i < 16; ++i) {
      int n = seg * 16 + i;
      s += x1_s[b][n] * v_y[n];
    }
    red[b][seg] = s;
  }
  __syncthreads();
  if (tid < 16) {
    float s = b_vy[0];
    for (int i = 0; i < 16; ++i) s += red[tid][i];
    out[b0 + tid] = s;
  }
}

extern "C" void kernel_launch(void* const* d_in, const int* in_sizes, int n_in,
                              void* d_out, int out_size, void* d_ws, size_t ws_size,
                              hipStream_t stream) {
  (void)in_sizes; (void)n_in; (void)out_size; (void)ws_size;
  const float* enc   = (const float*)d_in[0];
  const float* yin   = (const float*)d_in[1];
  const float* W_d   = (const float*)d_in[2];
  const float* b_Wd  = (const float*)d_in[3];
  const float* U_d   = (const float*)d_in[4];
  const float* v_d   = (const float*)d_in[5];
  const float* w_til = (const float*)d_in[6];
  const float* b_wt  = (const float*)d_in[7];
  const float* W_ih  = (const float*)d_in[8];
  const float* W_hh  = (const float*)d_in[9];
  const float* b_ih  = (const float*)d_in[10];
  const float* b_hh  = (const float*)d_in[11];
  const float* W_y   = (const float*)d_in[12];
  const float* b_Wy  = (const float*)d_in[13];
  const float* v_y   = (const float*)d_in[14];
  const float* b_vy  = (const float*)d_in[15];

  char* ws = (char*)d_ws;
  size_t off = 0;
  __bf16* enc_bf = (__bf16*)(ws + off); off += (size_t)B_DIM * T_DIM * M_DIM * 2;
  float*  y1     = (float*)(ws + off);  off += (size_t)B_DIM * T_DIM * M_DIM * 4;
  __bf16* Ud_p   = (__bf16*)(ws + off); off += (size_t)M_DIM * M_DIM * 2;
  __bf16* Wd_p   = (__bf16*)(ws + off); off += (size_t)M_DIM * 2 * H_DIM * 2;
  __bf16* Whh_p  = (__bf16*)(ws + off); off += (size_t)4 * H_DIM * H_DIM * 2;
  __bf16* Wy_p   = (__bf16*)(ws + off); off += (size_t)H_DIM * (H_DIM + M_DIM) * 2;

  enc_convert_kernel<<<(B_DIM * T_DIM * M_DIM) / 256, 256, 0, stream>>>(enc, enc_bf);
  pack_b_kernel<<<(M_DIM * M_DIM) / 256, 256, 0, stream>>>(U_d, Ud_p, M_DIM, M_DIM);
  pack_b_kernel<<<(M_DIM * 2 * H_DIM) / 256, 256, 0, stream>>>(W_d, Wd_p, M_DIM, 2 * H_DIM);
  pack_b_kernel<<<(4 * H_DIM * H_DIM) / 256, 256, 0, stream>>>(W_hh, Whh_p, 4 * H_DIM, H_DIM);
  pack_b_kernel<<<(H_DIM * (H_DIM + M_DIM)) / 256, 256, 0, stream>>>(W_y, Wy_p, H_DIM,
                                                                     H_DIM + M_DIM);
  const int tiles = (B_DIM * T_DIM / 16) * (M_DIM / 16);   // 16384 wave-tiles
  y1_gemm_kernel<<<tiles / 8, 256, 0, stream>>>(enc_bf, Ud_p, y1);

  recurrence_kernel<<<B_DIM / 16, 512, 0, stream>>>(
      enc, yin, y1, Wd_p, Whh_p, Wy_p, b_Wd, v_d, w_til, b_wt, W_ih, b_ih, b_hh,
      b_Wy, v_y, b_vy, (float*)d_out);
}
